// Llama4TextExperts_12627203850415
// MI455X (gfx1250) — compile-verified, matrix operand order checked by reading, not run on hardware
//
#include <hip/hip_runtime.h>
#include <hip/hip_bf16.h>

typedef __bf16 bf16_t;
typedef __attribute__((ext_vector_type(16))) __bf16 v16bf;
typedef __attribute__((ext_vector_type(8)))  __bf16 v8bf;
typedef __attribute__((ext_vector_type(4)))  __bf16 v4bf;
typedef __attribute__((ext_vector_type(2)))  __bf16 v2bf;
typedef __attribute__((ext_vector_type(8)))  float  v8f;

#define NE 8
#define NT 1024
#define NH 4096
#define NF 4096

#define BM 128
#define BK 32

static __device__ __forceinline__ v8f wmma_bf16(v16bf a, v16bf b, v8f c) {
  // D = A(16x32) * B(32x16) + C, f32 accumulate
  return __builtin_amdgcn_wmma_f32_16x16x32_bf16(false, a, false, b, (short)0, c,
                                                 false, false);
}

// Load one 16x32 bf16 fragment from an LDS tile stored row-major with row
// stride BK. 'p0' = &tile[row][khalf] where khalf = (lane>=16 ? 8 : 0).
// VGPRs 0..3 <- K = khalf..khalf+7 ; VGPRs 4..7 <- K = khalf+16..khalf+23.
static __device__ __forceinline__ v16bf frag_ld(const bf16_t* p0) {
  v8bf lo = *(const v8bf*)(p0);
  v8bf hi = *(const v8bf*)(p0 + 16);
  return __builtin_shufflevector(lo, hi, 0, 1, 2, 3, 4, 5, 6, 7,
                                 8, 9, 10, 11, 12, 13, 14, 15);
}

// ---------------------------------------------------------------------------
// Kernel 1: inter[e,t,f] = up * silu(gate), gate_up = X @ W_gu  (fused)
// grid = (NF/64, NT/BM, NE), block = 256
// ---------------------------------------------------------------------------
__global__ __launch_bounds__(256) void moe_gateup_swiglu(
    const float* __restrict__ X, const float* __restrict__ Wgu,
    bf16_t* __restrict__ inter) {
  __shared__ bf16_t As[BM][BK];   // 8 KB  X tile (M x K)
  __shared__ bf16_t Bg[64][BK];   // 4 KB  gate weights, N-major (N x K)
  __shared__ bf16_t Bu[64][BK];   // 4 KB  up weights,   N-major (N x K)

  const int e    = blockIdx.z;
  const int m0   = blockIdx.y * BM;
  const int n0   = blockIdx.x * 64;
  const int tid  = threadIdx.x;
  const int lane = tid & 31;
  const int wave = tid >> 5;
  const int mw   = wave & 3;             // 4 wave-rows of 32 M
  const int nw   = wave >> 2;            // 2 wave-cols of 32 N
  const int khalf = (lane >> 4) << 3;    // 0 or 8
  const int l15   = lane & 15;

  const float* Xe = X + ((size_t)e * NT + m0) * NH;
  const float* Wg = Wgu + (size_t)e * NH * (2 * NF) + n0;

  const v8f zero = {0.f, 0.f, 0.f, 0.f, 0.f, 0.f, 0.f, 0.f};
  v8f accg[2][2], accu[2][2];
#pragma unroll
  for (int mi = 0; mi < 2; ++mi)
#pragma unroll
    for (int ni = 0; ni < 2; ++ni) { accg[mi][ni] = zero; accu[mi][ni] = zero; }

  // Per-thread staging coordinates (loop invariant)
  const int ar = tid >> 3;         // A row group
  const int ac = (tid & 7) * 4;    // A col (K)
  const int bk = (tid >> 4) * 2;   // B even k-row (0..30)
  const int bc = (tid & 15) * 4;   // B col (N)

  for (int k0 = 0; k0 < NH; k0 += BK) {
    // Stage A: 128x32 f32 -> bf16, one ds_store_b64 per float4
#pragma unroll
    for (int i = 0; i < 4; ++i) {
      int r = ar + i * 32;
      float4 v = *(const float4*)(Xe + (size_t)r * NH + (k0 + ac));
      v4bf t = {(bf16_t)v.x, (bf16_t)v.y, (bf16_t)v.z, (bf16_t)v.w};
      *(v4bf*)&As[r][ac] = t;
    }
    // Stage B gate+up: two adjacent k-rows per thread -> packed (k,k+1)
    // bf16 pairs -> one ds_store_b32 per n in the N-major tile.
    {
      const float* pg0 = Wg + (size_t)(k0 + bk) * (2 * NF) + bc;
      const float* pg1 = pg0 + (2 * NF);
      float4 g0 = *(const float4*)pg0;
      float4 g1 = *(const float4*)pg1;
      float4 u0 = *(const float4*)(pg0 + NF);
      float4 u1 = *(const float4*)(pg1 + NF);
      v2bf tg0 = {(bf16_t)g0.x, (bf16_t)g1.x};
      v2bf tg1 = {(bf16_t)g0.y, (bf16_t)g1.y};
      v2bf tg2 = {(bf16_t)g0.z, (bf16_t)g1.z};
      v2bf tg3 = {(bf16_t)g0.w, (bf16_t)g1.w};
      v2bf tu0 = {(bf16_t)u0.x, (bf16_t)u1.x};
      v2bf tu1 = {(bf16_t)u0.y, (bf16_t)u1.y};
      v2bf tu2 = {(bf16_t)u0.z, (bf16_t)u1.z};
      v2bf tu3 = {(bf16_t)u0.w, (bf16_t)u1.w};
      *(v2bf*)&Bg[bc + 0][bk] = tg0;
      *(v2bf*)&Bg[bc + 1][bk] = tg1;
      *(v2bf*)&Bg[bc + 2][bk] = tg2;
      *(v2bf*)&Bg[bc + 3][bk] = tg3;
      *(v2bf*)&Bu[bc + 0][bk] = tu0;
      *(v2bf*)&Bu[bc + 1][bk] = tu1;
      *(v2bf*)&Bu[bc + 2][bk] = tu2;
      *(v2bf*)&Bu[bc + 3][bk] = tu3;
    }
    __syncthreads();

    v16bf a[2], bg[2], bu[2];
#pragma unroll
    for (int mi = 0; mi < 2; ++mi)
      a[mi] = frag_ld(&As[mw * 32 + mi * 16 + l15][khalf]);
#pragma unroll
    for (int ni = 0; ni < 2; ++ni) {
      bg[ni] = frag_ld(&Bg[nw * 32 + ni * 16 + l15][khalf]);
      bu[ni] = frag_ld(&Bu[nw * 32 + ni * 16 + l15][khalf]);
    }
#pragma unroll
    for (int mi = 0; mi < 2; ++mi)
#pragma unroll
      for (int ni = 0; ni < 2; ++ni) {
        accg[mi][ni] = wmma_bf16(a[mi], bg[ni], accg[mi][ni]);
        accu[mi][ni] = wmma_bf16(a[mi], bu[ni], accu[mi][ni]);
      }
    __syncthreads();
  }

  // SwiGLU on f32 accumulators, emit bf16 intermediate.
  bf16_t* outp = inter + (size_t)e * NT * NF;
#pragma unroll
  for (int mi = 0; mi < 2; ++mi) {
    int mbase = m0 + mw * 32 + mi * 16 + khalf;  // khalf doubles as +8 row shift
#pragma unroll
    for (int ni = 0; ni < 2; ++ni) {
      int n = n0 + nw * 32 + ni * 16 + l15;
#pragma unroll
      for (int r = 0; r < 8; ++r) {
        float g = accg[mi][ni][r];
        float u = accu[mi][ni][r];
        // silu(g) = g * sigmoid(g); denominator >= 1 so fast v_rcp_f32 is safe
        float s = g * __builtin_amdgcn_rcpf(1.0f + __expf(-g));
        outp[(size_t)(mbase + r) * NF + n] = (bf16_t)(u * s);
      }
    }
  }
}

// ---------------------------------------------------------------------------
// Kernel 2: out[e,t,h] = inter @ W_d
// grid = (NH/128, NT/BM, NE), block = 256
// ---------------------------------------------------------------------------
__global__ __launch_bounds__(256) void moe_down(
    const bf16_t* __restrict__ inter, const float* __restrict__ Wd,
    float* __restrict__ out) {
  __shared__ bf16_t As[BM][BK];    // 8 KB  inter tile (M x K), already bf16
  __shared__ bf16_t Bs[128][BK];   // 8 KB  down weights, N-major (N x K)

  const int e    = blockIdx.z;
  const int m0   = blockIdx.y * BM;
  const int n0   = blockIdx.x * 128;
  const int tid  = threadIdx.x;
  const int lane = tid & 31;
  const int wave = tid >> 5;
  const int mw   = wave & 3;             // 4 wave-rows of 32 M
  const int nw   = wave >> 2;            // 2 wave-cols of 64 N
  const int khalf = (lane >> 4) << 3;
  const int l15   = lane & 15;

  const bf16_t* Ae = inter + ((size_t)e * NT + m0) * NF;
  const float*  We = Wd + (size_t)e * NF * NH + n0;

  const v8f zero = {0.f, 0.f, 0.f, 0.f, 0.f, 0.f, 0.f, 0.f};
  v8f acc[2][4];
#pragma unroll
  for (int mi = 0; mi < 2; ++mi)
#pragma unroll
    for (int ni = 0; ni < 4; ++ni) acc[mi][ni] = zero;

  for (int k0 = 0; k0 < NF; k0 += BK) {
    // Stage A: 128x32 bf16, straight 16-byte copies
#pragma unroll
    for (int i = 0; i < 2; ++i) {
      int idx = tid + i * 256;
      int r = idx >> 2;
      int c = (idx & 3) * 8;
      *(v8bf*)&As[r][c] = *(const v8bf*)(Ae + (size_t)r * NF + (k0 + c));
    }
    // Stage B: 32x128 f32, two adjacent k-rows per thread -> packed pairs
#pragma unroll
    for (int i = 0; i < 2; ++i) {
      int idx = tid + i * 256;          // 0..511
      int kr = (idx >> 5) * 2;          // even k-row 0..30
      int c  = (idx & 31) * 4;          // n col 0..124
      const float* p0 = We + (size_t)(k0 + kr) * NH + c;
      float4 a0 = *(const float4*)p0;
      float4 a1 = *(const float4*)(p0 + NH);
      v2bf t0 = {(bf16_t)a0.x, (bf16_t)a1.x};
      v2bf t1 = {(bf16_t)a0.y, (bf16_t)a1.y};
      v2bf t2 = {(bf16_t)a0.z, (bf16_t)a1.z};
      v2bf t3 = {(bf16_t)a0.w, (bf16_t)a1.w};
      *(v2bf*)&Bs[c + 0][kr] = t0;
      *(v2bf*)&Bs[c + 1][kr] = t1;
      *(v2bf*)&Bs[c + 2][kr] = t2;
      *(v2bf*)&Bs[c + 3][kr] = t3;
    }
    __syncthreads();

    v16bf a[2], b[4];
#pragma unroll
    for (int mi = 0; mi < 2; ++mi)
      a[mi] = frag_ld(&As[mw * 32 + mi * 16 + l15][khalf]);
#pragma unroll
    for (int ni = 0; ni < 4; ++ni)
      b[ni] = frag_ld(&Bs[nw * 64 + ni * 16 + l15][khalf]);
#pragma unroll
    for (int mi = 0; mi < 2; ++mi)
#pragma unroll
      for (int ni = 0; ni < 4; ++ni)
        acc[mi][ni] = wmma_bf16(a[mi], b[ni], acc[mi][ni]);
    __syncthreads();
  }

#pragma unroll
  for (int mi = 0; mi < 2; ++mi) {
    int mbase = m0 + mw * 32 + mi * 16 + khalf;
#pragma unroll
    for (int ni = 0; ni < 4; ++ni) {
      int n = n0 + nw * 64 + ni * 16 + l15;
#pragma unroll
      for (int r = 0; r < 8; ++r)
        out[((size_t)e * NT + mbase + r) * NH + n] = acc[mi][ni][r];
    }
  }
}

extern "C" void kernel_launch(void* const* d_in, const int* in_sizes, int n_in,
                              void* d_out, int out_size, void* d_ws, size_t ws_size,
                              hipStream_t stream) {
  (void)in_sizes; (void)n_in; (void)out_size; (void)ws_size;
  const float* X   = (const float*)d_in[0];   // (E*T, H) f32
  const float* Wgu = (const float*)d_in[1];   // (E, H, 2F) f32
  const float* Wd  = (const float*)d_in[2];   // (E, F, H) f32
  float*  out   = (float*)d_out;              // (E*T, H) f32
  bf16_t* inter = (bf16_t*)d_ws;              // (E, T, F) bf16, 64 MB scratch

  dim3 g1(NF / 64, NT / BM, NE);
  moe_gateup_swiglu<<<g1, 256, 0, stream>>>(X, Wgu, inter);

  dim3 g2(NH / 128, NT / BM, NE);
  moe_down<<<g2, 256, 0, stream>>>(inter, Wd, out);
}
